// HedgerOffloadingPPO_23055384445711
// MI455X (gfx1250) — matrix operand order repaired and checked by reading, not compiled, not run on hardware
//
#include <hip/hip_runtime.h>
#include <hip/hip_bf16.h>
#include <math.h>

// ---------------------------------------------------------------------------
// Problem constants (match reference)
// ---------------------------------------------------------------------------
#define DD   512        // d_model
#define FF   128        // feature dim
#define MSN  1024       // logic nodes
#define NPN  16384      // physical nodes
#define ELN  4096       // logic edges
#define EPN  131072     // phys edges
#define SRCN 0
#define NEGV (-1.0e9f)

typedef __attribute__((ext_vector_type(16))) _Float16 v16h;
typedef __attribute__((ext_vector_type(8)))  _Float16 v8h;
typedef __attribute__((ext_vector_type(8)))  float    v8f;

// ---------------------------------------------------------------------------
// Tiled WMMA GEMM: C[M,N] = post( (A (+A2)) @ B ), f32 in/out, f16 WMMA.
// Block tile 64x256, 256 threads = 8 waves (2 in M x 4 in N); each wave owns a
// 32x64 region via 2x4 v_wmma_f32_16x16x32_f16 fragments (8 WMMAs/k-step).
// Template params make staging branch-free; software-pipelined global->reg
// prefetch overlaps the next tile's loads with the current tile's WMMAs.
// ---------------------------------------------------------------------------
#define TM  64
#define TN  256
#define TKG 32

union H4 { _Float16 h[4]; unsigned long long u; };

static __device__ __forceinline__ unsigned long long pack4(float4 v) {
  H4 p;
  p.h[0] = (_Float16)v.x; p.h[1] = (_Float16)v.y;
  p.h[2] = (_Float16)v.z; p.h[3] = (_Float16)v.w;
  return p.u;
}

template <int ADDA, int TRANSB>
__global__ __launch_bounds__(256) void gemm_wmma(
    const float* __restrict__ A, const float* __restrict__ A2,
    const float* __restrict__ B, float* __restrict__ C,
    int M, int N, int K, int doRelu, int accum, float scale)
{
  __shared__ __attribute__((aligned(64))) _Float16 As[TM][TKG];  // [m][k]
  __shared__ __attribute__((aligned(64))) _Float16 Bs[TN][TKG];  // [n][k]

  const int tid  = threadIdx.x;
  const int lane = tid & 31;
  const int wid  = tid >> 5;
  const int m0   = (wid & 1) * 32;   // 2 waves along M (32 rows each)
  const int n0   = (wid >> 1) * 64;  // 4 waves along N (64 cols each)
  const int blockM = blockIdx.y * TM;
  const int blockN = blockIdx.x * TN;
  const int l15     = lane & 15;
  const int halfSel = lane >> 4;     // 0 or 1

  // ---- fixed per-thread staging coordinates ----
  // A: 64x32 floats = 512 float4 -> 2 per thread
  int aR[2], aC[2];
  for (int e = 0; e < 2; ++e) {
    int idx = tid + (e << 8);
    aR[e] = idx >> 3;            // 0..63
    aC[e] = (idx & 7) << 2;      // 0,4,..,28
  }
  // B: 256x32 halves = 2048 float4 of source -> 8 per thread
  int bX[8], bY[8];
  for (int e = 0; e < 8; ++e) {
    int idx = tid + (e << 8);
    if (TRANSB) { bX[e] = idx >> 3;  bY[e] = (idx & 7) << 2; }   // n (0..255), k
    else        { bX[e] = idx >> 6;  bY[e] = (idx & 63) << 2; }  // k (0..31),  n
  }

  v8f acc[2][4];
  for (int a = 0; a < 2; ++a)
    for (int b = 0; b < 4; ++b)
      for (int e = 0; e < 8; ++e) acc[a][b][e] = 0.0f;

  float4 aReg[2], bReg[8];

  // ---- prologue: load k-tile 0 into registers ----
  for (int e = 0; e < 2; ++e) {
    size_t off = (size_t)(blockM + aR[e]) * K + aC[e];
    aReg[e] = *(const float4*)(A + off);
    if (ADDA) {
      float4 t = *(const float4*)(A2 + off);
      aReg[e].x += t.x; aReg[e].y += t.y; aReg[e].z += t.z; aReg[e].w += t.w;
    }
  }
  for (int e = 0; e < 8; ++e) {
    size_t off = TRANSB ? (size_t)(blockN + bX[e]) * K + bY[e]
                        : (size_t)bX[e] * N + blockN + bY[e];
    bReg[e] = *(const float4*)(B + off);
  }

  for (int kb = 0; kb < K; kb += TKG) {
    // ---- commit registers -> LDS ----
    for (int e = 0; e < 2; ++e)
      *(unsigned long long*)&As[aR[e]][aC[e]] = pack4(aReg[e]);
    if (TRANSB) {
      for (int e = 0; e < 8; ++e)
        *(unsigned long long*)&Bs[bX[e]][bY[e]] = pack4(bReg[e]);
    } else {
      for (int e = 0; e < 8; ++e) {
        Bs[bY[e] + 0][bX[e]] = (_Float16)bReg[e].x;
        Bs[bY[e] + 1][bX[e]] = (_Float16)bReg[e].y;
        Bs[bY[e] + 2][bX[e]] = (_Float16)bReg[e].z;
        Bs[bY[e] + 3][bX[e]] = (_Float16)bReg[e].w;
      }
    }
    __syncthreads();

    // ---- issue next tile's global loads (in flight during WMMAs) ----
    int kn = kb + TKG;
    if (kn < K) {
      for (int e = 0; e < 2; ++e) {
        size_t off = (size_t)(blockM + aR[e]) * K + kn + aC[e];
        aReg[e] = *(const float4*)(A + off);
        if (ADDA) {
          float4 t = *(const float4*)(A2 + off);
          aReg[e].x += t.x; aReg[e].y += t.y; aReg[e].z += t.z; aReg[e].w += t.w;
        }
      }
      for (int e = 0; e < 8; ++e) {
        size_t off = TRANSB ? (size_t)(blockN + bX[e]) * K + kn + bY[e]
                            : (size_t)(kn + bX[e]) * N + blockN + bY[e];
        bReg[e] = *(const float4*)(B + off);
      }
      // prefetch the tile after next into L2 (global_prefetch_b8)
      int kp = kn + TKG;
      if (kp < K) {
        __builtin_prefetch(A + (size_t)(blockM + aR[0]) * K + kp + aC[0], 0, 1);
        __builtin_prefetch(TRANSB ? B + (size_t)(blockN + bX[0]) * K + kp + bY[0]
                                  : B + (size_t)(kp + bX[0]) * N + blockN + bY[0],
                           0, 1);
      }
    }

    // ---- fragment loads (contiguous LDS runs; k = 16*half + s in both A and B) ----
    // A 16x32 f16: lanes 0-15 hold K=0..15; lanes 16-31 K=16..31
    v16h afr[2], bfr[4];
    for (int im = 0; im < 2; ++im) {
      int row = m0 + im * 16 + l15;
      int kb0 = halfSel * 16;
      afr[im] = *(const v16h*)&As[row][kb0];
    }
    // B 32x16 f16: lanes 0-15 col n hold K=0..15; lanes 16-31 K=16..31
    for (int in = 0; in < 4; ++in) {
      int coln = n0 + in * 16 + l15;
      int kb0  = halfSel * 16;
      bfr[in] = *(const v16h*)&Bs[coln][kb0];
    }
    for (int im = 0; im < 2; ++im)
      for (int in = 0; in < 4; ++in)
        acc[im][in] = __builtin_amdgcn_wmma_f32_16x16x32_f16(
            false, afr[im], false, bfr[in], (short)0, acc[im][in], false, false);
    __syncthreads();
  }

  // ---- epilogue: C/D layout: VGPR r -> M=r (lanes 0-15) / r+8 (16-31), N=lane&15
  for (int im = 0; im < 2; ++im)
    for (int in = 0; in < 4; ++in)
      for (int r = 0; r < 8; ++r) {
        int m = blockM + m0 + im * 16 + r + (halfSel ? 8 : 0);
        int n = blockN + n0 + in * 16 + l15;
        size_t off = (size_t)m * N + n;
        float v = acc[im][in][r] * scale;
        if (accum)  v += C[off];
        if (doRelu) v = fmaxf(v, 0.0f);
        C[off] = v;
      }
}

// ---------------------------------------------------------------------------
// Utility kernels
// ---------------------------------------------------------------------------
__global__ void zerof_k(float* p, int n) {
  int i = blockIdx.x * blockDim.x + threadIdx.x;
  if (i < n) p[i] = 0.0f;
}
__global__ void filli_k(int* p, int v, int n) {
  int i = blockIdx.x * blockDim.x + threadIdx.x;
  if (i < n) p[i] = v;
}
__global__ void init_dist_k(int* d, int n, int src, int big) {
  int i = blockIdx.x * blockDim.x + threadIdx.x;
  if (i < n) d[i] = (i == src) ? 0 : big;
}
__global__ void copyi_k(int* dst, const int* src, int n) {
  int i = blockIdx.x * blockDim.x + threadIdx.x;
  if (i < n) dst[i] = src[i];
}

// segment_sum: agg[col[e]] += h[row[e]], one block (128 thr * float4) per edge
__global__ __launch_bounds__(128) void scatter_add_k(
    const int* __restrict__ ei, const float* __restrict__ h,
    float* __restrict__ agg, int ne)
{
  int e = blockIdx.x;
  int r = ei[e], c = ei[ne + e];
  const float4* src = (const float4*)(h + (size_t)r * DD);
  float4 v = src[threadIdx.x];
  float* dst = agg + (size_t)c * DD + threadIdx.x * 4;
  atomicAdd(dst + 0, v.x); atomicAdd(dst + 1, v.y);
  atomicAdd(dst + 2, v.z); atomicAdd(dst + 3, v.w);
}

// one Bellman-Ford relaxation: next[col] = min(next[col], min(dist[row]+1,big))
__global__ void bfs_relax_k(const int* __restrict__ ei,
                            const int* __restrict__ dist,
                            int* __restrict__ next, int ne, int big)
{
  int e = blockIdx.x * blockDim.x + threadIdx.x;
  if (e >= ne) return;
  int cand = dist[ei[e]] + 1;
  if (cand > big) cand = big;
  atomicMin(&next[ei[ne + e]], cand);
}

// sequential state trajectory: last[t], first-visit time per node.
__global__ void trajectory_k(const int* __restrict__ topo,
                             const int* __restrict__ acts,
                             int* __restrict__ last_arr,
                             int* __restrict__ visit_time, int steps)
{
  if (threadIdx.x || blockIdx.x) return;
  int last = SRCN;
  for (int t = 0; t < steps; ++t) {
    int i = topo[t];
    last_arr[t] = last;
    int a = acts[i];
    if (a != last) {
      if (visit_time[last] == 0x7fffffff) visit_time[last] = t;
      last = a;
    }
  }
}

// one block per scan step: masked log-softmax over NPN nodes, accumulate
// logp_sum (out[0]) and ent_sum (out[1]).
__global__ __launch_bounds__(256) void step_softmax_k(
    const float* __restrict__ logits, const unsigned char* __restrict__ mask,
    const int* __restrict__ metric, const int* __restrict__ visit_time,
    const int* __restrict__ last_arr, const int* __restrict__ topo,
    const int* __restrict__ acts, float* __restrict__ out)
{
  const int t = blockIdx.x, tid = threadIdx.x;
  const int i = topo[t];
  const int last = last_arr[t];
  const int cloud = NPN - 1;
  const int mlast = metric[last];
  const unsigned char* base = mask + (size_t)i * NPN;
  const float* L = logits + (size_t)t * NPN;
  const bool lastCloud = (last == cloud);
  const bool baseCloud = base[cloud] != 0;

  unsigned long long bits = 0ull;
  for (int j = 0; j < 64; ++j) {
    int n = tid + (j << 8);
    bool allowed;
    if (lastCloud) {
      allowed = (n == cloud) && baseCloud;
    } else {
      bool b = base[n] != 0;
      bool forbid = (visit_time[n] < t) && (n != last);
      allowed = b && !forbid && (metric[n] >= mlast);
    }
    bits |= ((unsigned long long)(allowed ? 1u : 0u)) << j;
  }

  __shared__ int s_any;
  __shared__ float red[256];
  if (tid == 0) s_any = 0;
  __syncthreads();
  if (bits) s_any = 1;
  __syncthreads();
  const bool any_a = (s_any != 0);

  if (!any_a) {  // relaxation fallback: stay -> cloud -> base
    bool baseLast = base[last] != 0;
    bits = 0ull;
    for (int j = 0; j < 64; ++j) {
      int n = tid + (j << 8);
      bool allowed = baseLast ? (n == last)
                              : (baseCloud ? (n == cloud) : (base[n] != 0));
      bits |= ((unsigned long long)(allowed ? 1u : 0u)) << j;
    }
  }

  // max
  float mx = -3.0e38f;
  for (int j = 0; j < 64; ++j) {
    int n = tid + (j << 8);
    float x = ((bits >> j) & 1ull) ? L[n] : NEGV;
    mx = fmaxf(mx, x);
  }
  red[tid] = mx; __syncthreads();
  for (int s = 128; s; s >>= 1) {
    if (tid < s) red[tid] = fmaxf(red[tid], red[tid + s]);
    __syncthreads();
  }
  const float m = red[0];
  __syncthreads();

  // sum exp, sum d*exp
  float s1 = 0.0f, s2 = 0.0f;
  for (int j = 0; j < 64; ++j) {
    int n = tid + (j << 8);
    float x = ((bits >> j) & 1ull) ? L[n] : NEGV;
    float d = x - m;
    float e = __expf(d);
    s1 += e; s2 += d * e;
  }
  red[tid] = s1; __syncthreads();
  for (int s = 128; s; s >>= 1) { if (tid < s) red[tid] += red[tid + s]; __syncthreads(); }
  const float S1 = red[0]; __syncthreads();
  red[tid] = s2; __syncthreads();
  for (int s = 128; s; s >>= 1) { if (tid < s) red[tid] += red[tid + s]; __syncthreads(); }
  const float S2 = red[0]; __syncthreads();

  if (tid == 0) {
    float logS = logf(S1);
    int a = acts[i];
    bool aa;
    if (any_a) {
      if (lastCloud) aa = (a == cloud) && baseCloud;
      else {
        bool b = base[a] != 0;
        bool forbid = (visit_time[a] < t) && (a != last);
        aa = b && !forbid && (metric[a] >= mlast);
      }
    } else {
      bool baseLast = base[last] != 0;
      aa = baseLast ? (a == last) : (baseCloud ? (a == cloud) : (base[a] != 0));
    }
    float xa = aa ? L[a] : NEGV;
    atomicAdd(&out[0], xa - m - logS);
    atomicAdd(&out[1], logS - S2 / S1);   // entropy
  }
}

// column partial sums (for mean pooling), 512 threads = one column each
__global__ __launch_bounds__(512) void pool_cols_k(const float* __restrict__ h,
                                                   float* __restrict__ pooled,
                                                   int rowsPerBlock)
{
  int d = threadIdx.x;
  int r0 = blockIdx.x * rowsPerBlock;
  float acc = 0.0f;
  for (int r = 0; r < rowsPerBlock; ++r) acc += h[(size_t)(r0 + r) * DD + d];
  atomicAdd(&pooled[d], acc);
}

// value head: relu(pooled/Mk @ Wc1 + bc1) @ wc2 -> out[2]
__global__ __launch_bounds__(512) void value_head_k(
    const float* __restrict__ pooled, const float* __restrict__ Wc1,
    const float* __restrict__ bc1, const float* __restrict__ wc2,
    float* __restrict__ out)
{
  int j = threadIdx.x;
  float acc = bc1[j];
  for (int k = 0; k < 2 * DD; ++k) {
    float p = pooled[k] * ((k < DD) ? (1.0f / (float)MSN) : (1.0f / (float)NPN));
    acc += p * Wc1[(size_t)k * DD + j];
  }
  float v = fmaxf(acc, 0.0f) * wc2[j];
  __shared__ float red[512];
  red[j] = v; __syncthreads();
  for (int s = 256; s; s >>= 1) { if (j < s) red[j] += red[j + s]; __syncthreads(); }
  if (j == 0) out[2] = red[0];
}

// ---------------------------------------------------------------------------
// Host side
// ---------------------------------------------------------------------------
static inline void launch_gemm(const float* A, const float* A2, const float* B,
                               float* C, int M, int N, int K, int transB,
                               int relu, int accum, float scale, hipStream_t s)
{
  dim3 grid(N / TN, M / TM);
  if (A2)
    gemm_wmma<1, 0><<<grid, 256, 0, s>>>(A, A2, B, C, M, N, K, relu, accum, scale);
  else if (transB)
    gemm_wmma<0, 1><<<grid, 256, 0, s>>>(A, nullptr, B, C, M, N, K, relu, accum, scale);
  else
    gemm_wmma<0, 0><<<grid, 256, 0, s>>>(A, nullptr, B, C, M, N, K, relu, accum, scale);
}

extern "C" void kernel_launch(void* const* d_in, const int* in_sizes, int n_in,
                              void* d_out, int out_size, void* d_ws, size_t ws_size,
                              hipStream_t stream)
{
  const int*   lEdges  = (const int*)  d_in[0];
  const float* lFeats  = (const float*)d_in[1];
  const int*   pEdges  = (const int*)  d_in[2];
  const float* pFeats  = (const float*)d_in[3];
  const int*   actions = (const int*)  d_in[4];
  const unsigned char* staticMask = (const unsigned char*)d_in[5];
  const int*   topo    = (const int*)  d_in[6];
  const float* WsIn  = (const float*)d_in[7];
  const float* WsOut = (const float*)d_in[8];
  const float* WpIn  = (const float*)d_in[9];
  const float* WpOut = (const float*)d_in[10];
  const float* sA1   = (const float*)d_in[11];
  const float* sA2   = (const float*)d_in[12];
  const float* pA1   = (const float*)d_in[13];
  const float* pA2   = (const float*)d_in[14];
  const float* Wq    = (const float*)d_in[15];
  const float* Wk    = (const float*)d_in[16];
  const float* Wc1   = (const float*)d_in[17];
  const float* bc1   = (const float*)d_in[18];
  const float* wc2   = (const float*)d_in[19];
  float* out = (float*)d_out;

  // ---- workspace carve-up ----
  char* w = (char*)d_ws;
  size_t o = 0;
  float* hpreS = (float*)(w + o); o += (size_t)MSN * DD * 4;   // 2 MB
  float* aggS  = (float*)(w + o); o += (size_t)MSN * DD * 4;
  float* hS    = (float*)(w + o); o += (size_t)MSN * DD * 4;
  float* hpreP = (float*)(w + o); o += (size_t)NPN * DD * 4;   // 32 MB
  float* aggP  = (float*)(w + o); o += (size_t)NPN * DD * 4;
  float* hP    = (float*)(w + o); o += (size_t)NPN * DD * 4;
  float* logitsBuf = (float*)(w + o); o += (size_t)MSN * NPN * 4;  // 64 MB
  int* dist0   = (int*)(w + o); o += (size_t)NPN * 4;
  int* dist1   = (int*)(w + o); o += (size_t)NPN * 4;
  int* visitT  = (int*)(w + o); o += (size_t)NPN * 4;
  int* lastArr = (int*)(w + o); o += (size_t)MSN * 4;
  float* pooled = (float*)(w + o); o += (size_t)2 * DD * 4;
  float* tS = hpreS;   // reuse after h_s computed
  float* qAll = aggS;
  float* tP = hpreP;
  float* Kmat = aggP;

  const float qscale = 1.0f / sqrtf((float)DD);

  // ---- init ----
  zerof_k<<<(MSN * DD + 255) / 256, 256, 0, stream>>>(aggS, MSN * DD);
  zerof_k<<<(NPN * DD + 255) / 256, 256, 0, stream>>>(aggP, NPN * DD);
  zerof_k<<<4, 256, 0, stream>>>(pooled, 2 * DD);
  zerof_k<<<1, 32, 0, stream>>>(out, 2);
  filli_k<<<(NPN + 255) / 256, 256, 0, stream>>>(visitT, 0x7fffffff, NPN);
  init_dist_k<<<(NPN + 255) / 256, 256, 0, stream>>>(dist0, NPN, SRCN, NPN);

  // ---- GNN stage 1: h_pre = relu(feats @ W_in) ----
  launch_gemm(lFeats, nullptr, WsIn, hpreS, MSN, DD, FF, 0, 1, 0, 1.0f, stream);
  launch_gemm(pFeats, nullptr, WpIn, hpreP, NPN, DD, FF, 0, 1, 0, 1.0f, stream);

  // ---- segment_sum aggregation ----
  scatter_add_k<<<ELN, 128, 0, stream>>>(lEdges, hpreS, aggS, ELN);
  scatter_add_k<<<EPN, 128, 0, stream>>>(pEdges, hpreP, aggP, EPN);

  // ---- GNN stage 2: h = relu((h_pre + agg) @ W_out) ----
  launch_gemm(hpreS, aggS, WsOut, hS, MSN, DD, DD, 0, 1, 0, 1.0f, stream);
  launch_gemm(hpreP, aggP, WpOut, hP, NPN, DD, DD, 0, 1, 0, 1.0f, stream);

  // ---- residual adapters: h += relu(h @ A1) @ A2 ----
  launch_gemm(hS, nullptr, sA1, tS, MSN, DD, DD, 0, 1, 0, 1.0f, stream);
  launch_gemm(tS, nullptr, sA2, hS, MSN, DD, DD, 0, 0, 1, 1.0f, stream);
  launch_gemm(hP, nullptr, pA1, tP, NPN, DD, DD, 0, 1, 0, 1.0f, stream);
  launch_gemm(tP, nullptr, pA2, hP, NPN, DD, DD, 0, 0, 1, 1.0f, stream);

  // ---- q/K projections, then full logits GEMM (q_all @ K^T) ----
  launch_gemm(hS, nullptr, Wq, qAll, MSN, DD, DD, 0, 0, 0, qscale, stream);
  launch_gemm(hP, nullptr, Wk, Kmat, NPN, DD, DD, 0, 0, 0, 1.0f, stream);
  launch_gemm(qAll, nullptr, Kmat, logitsBuf, MSN, NPN, DD, /*transB=*/1, 0, 0, 1.0f, stream);

  // ---- BFS hop metric: 16 synchronous Bellman-Ford relaxations ----
  int* cur = dist0;
  int* nxt = dist1;
  for (int it = 0; it < 16; ++it) {
    copyi_k<<<(NPN + 255) / 256, 256, 0, stream>>>(nxt, cur, NPN);
    bfs_relax_k<<<(EPN + 255) / 256, 256, 0, stream>>>(pEdges, cur, nxt, EPN, NPN);
    int* tmp = cur; cur = nxt; nxt = tmp;
  }

  // ---- scan state trajectory (cheap sequential), then parallel softmaxes ----
  trajectory_k<<<1, 1, 0, stream>>>(topo, actions, lastArr, visitT, MSN);
  step_softmax_k<<<MSN, 256, 0, stream>>>(logitsBuf, staticMask, cur, visitT,
                                          lastArr, topo, actions, out);

  // ---- value head ----
  pool_cols_k<<<MSN / 64, 512, 0, stream>>>(hS, pooled, 64);
  pool_cols_k<<<NPN / 64, 512, 0, stream>>>(hP, pooled + DD, 64);
  value_head_k<<<1, 512, 0, stream>>>(pooled, Wc1, bc1, wc2, out);
}
